// MHADecoderLayer_28638841929751
// MI455X (gfx1250) — matrix operand order repaired and checked by reading, not compiled, not run on hardware
//
#include <hip/hip_runtime.h>

typedef unsigned short bf16_t;
typedef __attribute__((ext_vector_type(16))) __bf16 v16bf;
typedef __attribute__((ext_vector_type(8)))  float  v8f;

union Frag {
  uint4  u[2];
  v16bf  v;
  bf16_t h[16];
};
static_assert(sizeof(Frag) == 32, "frag size");

__device__ __forceinline__ bf16_t f2bf(float f) {
  union { float f; unsigned int u; } x;
  x.f = f;
  unsigned int r = x.u + 0x7fffu + ((x.u >> 16) & 1u);
  return (bf16_t)(r >> 16);
}

constexpr int Bb  = 2;
constexpr int Ss  = 2048;
constexpr int Dm  = 2048;
constexpr int Hh  = 16;
constexpr int KVh = 4;
constexpr int Dh  = 128;
constexpr int Dff = 8192;
constexpr int Mtot = Bb * Ss;   // 4096 token rows

// ---------------------------------------------------------------------------
// Weight transpose f32 [K][N] -> bf16 [N][K]
// ---------------------------------------------------------------------------
__global__ void transpose_to_bf16(const float* __restrict__ src,
                                  bf16_t* __restrict__ dst, int K, int N) {
  __shared__ float tile[32][33];
  int kb = blockIdx.y * 32, nb = blockIdx.x * 32;
  int tx = threadIdx.x, ty = threadIdx.y;   // block 32x8
#pragma unroll
  for (int i = 0; i < 32; i += 8)
    tile[ty + i][tx] = src[(size_t)(kb + ty + i) * N + nb + tx];
  __syncthreads();
#pragma unroll
  for (int i = 0; i < 32; i += 8)
    dst[(size_t)(nb + ty + i) * K + kb + tx] = f2bf(tile[tx][ty + i]);
}

// ---------------------------------------------------------------------------
// RMSNorm f32 -> bf16 (one block per row, 256 threads)
// ---------------------------------------------------------------------------
__global__ __launch_bounds__(256)
void rmsnorm_bf16(const float* __restrict__ x, const float* __restrict__ w,
                  bf16_t* __restrict__ out, int D) {
  int row = blockIdx.x;
  const float* xr = x + (size_t)row * D;
  float ss = 0.f;
  for (int i = threadIdx.x; i < D; i += 256) { float v = xr[i]; ss += v * v; }
#pragma unroll
  for (int off = 16; off > 0; off >>= 1) ss += __shfl_xor(ss, off);
  __shared__ float red[8];
  int wid = threadIdx.x >> 5, lid = threadIdx.x & 31;
  if (lid == 0) red[wid] = ss;
  __syncthreads();
  if (wid == 0) {
    float v = (lid < 8) ? red[lid] : 0.f;
#pragma unroll
    for (int off = 4; off > 0; off >>= 1) v += __shfl_xor(v, off);
    if (lid == 0) red[0] = v;
  }
  __syncthreads();
  float rinv = rsqrtf(red[0] / (float)D + 1e-5f);
  for (int i = threadIdx.x; i < D; i += 256)
    out[(size_t)row * D + i] = f2bf(xr[i] * rinv * w[i]);
}

// ---------------------------------------------------------------------------
// Generic bf16 GEMM: C[M,N] = A[M,K] @ Bt[N,K]^T  (+ optional f32 residual)
// tile 128x128x32, 256 threads = 8 waves, each wave 32x64 (2x4 WMMA tiles)
// ---------------------------------------------------------------------------
#define LDSS 40   // LDS row stride in bf16 elems (80B, conflict-reducing)

__global__ __launch_bounds__(256)
void gemm_bf16(const bf16_t* __restrict__ A, const bf16_t* __restrict__ Bt,
               float* __restrict__ C, const float* __restrict__ resid,
               int M, int N, int K) {
  __shared__ bf16_t As[128 * LDSS];
  __shared__ bf16_t Bs[128 * LDSS];
  int bm = blockIdx.y * 128, bn = blockIdx.x * 128;
  int t = threadIdx.x;
  int wave = t >> 5, lane = t & 31, half = lane >> 4, l16 = lane & 15;
  int wm = (wave & 3) * 32, wn = (wave >> 2) * 64;

  v8f acc[2][4] = {};

  for (int k0 = 0; k0 < K; k0 += 32) {
#pragma unroll
    for (int i = 0; i < 2; i++) {
      int idx = t + i * 256;
      int row = idx >> 2, c8 = (idx & 3) * 8;
      *(uint4*)&As[row * LDSS + c8] =
          *(const uint4*)(A + (size_t)(bm + row) * K + k0 + c8);
      *(uint4*)&Bs[row * LDSS + c8] =
          *(const uint4*)(Bt + (size_t)(bn + row) * K + k0 + c8);
    }
    __syncthreads();
    Frag af[2], bf[4];
#pragma unroll
    for (int mi = 0; mi < 2; mi++) {
      int r = wm + mi * 16 + l16;
      af[mi].u[0] = *(const uint4*)&As[r * LDSS + half * 8];
      af[mi].u[1] = *(const uint4*)&As[r * LDSS + half * 8 + 16];
    }
#pragma unroll
    for (int ni = 0; ni < 4; ni++) {
      int r = wn + ni * 16 + l16;
      bf[ni].u[0] = *(const uint4*)&Bs[r * LDSS + half * 8];
      bf[ni].u[1] = *(const uint4*)&Bs[r * LDSS + half * 8 + 16];
    }
#pragma unroll
    for (int mi = 0; mi < 2; mi++)
#pragma unroll
      for (int ni = 0; ni < 4; ni++)
        acc[mi][ni] = __builtin_amdgcn_wmma_f32_16x16x32_bf16(
            false, af[mi].v, false, bf[ni].v, (short)0, acc[mi][ni], false, false);
    __syncthreads();
  }

#pragma unroll
  for (int mi = 0; mi < 2; mi++)
#pragma unroll
    for (int ni = 0; ni < 4; ni++)
#pragma unroll
      for (int r = 0; r < 8; r++) {
        int row = bm + wm + mi * 16 + half * 8 + r;
        int col = bn + wn + ni * 16 + l16;
        float v = acc[mi][ni][r];
        if (resid) v += resid[(size_t)row * N + col];
        C[(size_t)row * N + col] = v;
      }
}

// ---------------------------------------------------------------------------
// Fused gate/up GEMM: out = bf16( silu(A@Wg) * (A@Wu) ),  N = Dff
// ---------------------------------------------------------------------------
__global__ __launch_bounds__(256)
void gemm_gateup(const bf16_t* __restrict__ A, const bf16_t* __restrict__ Gt,
                 const bf16_t* __restrict__ Ut, bf16_t* __restrict__ out,
                 int M, int N, int K) {
  __shared__ bf16_t As[128 * LDSS];
  __shared__ bf16_t Gs[128 * LDSS];
  __shared__ bf16_t Us[128 * LDSS];
  int bm = blockIdx.y * 128, bn = blockIdx.x * 128;
  int t = threadIdx.x;
  int wave = t >> 5, lane = t & 31, half = lane >> 4, l16 = lane & 15;
  int wm = (wave & 3) * 32, wn = (wave >> 2) * 64;

  v8f accg[2][4] = {};
  v8f accu[2][4] = {};

  for (int k0 = 0; k0 < K; k0 += 32) {
#pragma unroll
    for (int i = 0; i < 2; i++) {
      int idx = t + i * 256;
      int row = idx >> 2, c8 = (idx & 3) * 8;
      *(uint4*)&As[row * LDSS + c8] =
          *(const uint4*)(A + (size_t)(bm + row) * K + k0 + c8);
      *(uint4*)&Gs[row * LDSS + c8] =
          *(const uint4*)(Gt + (size_t)(bn + row) * K + k0 + c8);
      *(uint4*)&Us[row * LDSS + c8] =
          *(const uint4*)(Ut + (size_t)(bn + row) * K + k0 + c8);
    }
    __syncthreads();
    Frag af[2], gf[4], uf[4];
#pragma unroll
    for (int mi = 0; mi < 2; mi++) {
      int r = wm + mi * 16 + l16;
      af[mi].u[0] = *(const uint4*)&As[r * LDSS + half * 8];
      af[mi].u[1] = *(const uint4*)&As[r * LDSS + half * 8 + 16];
    }
#pragma unroll
    for (int ni = 0; ni < 4; ni++) {
      int r = wn + ni * 16 + l16;
      gf[ni].u[0] = *(const uint4*)&Gs[r * LDSS + half * 8];
      gf[ni].u[1] = *(const uint4*)&Gs[r * LDSS + half * 8 + 16];
      uf[ni].u[0] = *(const uint4*)&Us[r * LDSS + half * 8];
      uf[ni].u[1] = *(const uint4*)&Us[r * LDSS + half * 8 + 16];
    }
#pragma unroll
    for (int mi = 0; mi < 2; mi++)
#pragma unroll
      for (int ni = 0; ni < 4; ni++) {
        accg[mi][ni] = __builtin_amdgcn_wmma_f32_16x16x32_bf16(
            false, af[mi].v, false, gf[ni].v, (short)0, accg[mi][ni], false, false);
        accu[mi][ni] = __builtin_amdgcn_wmma_f32_16x16x32_bf16(
            false, af[mi].v, false, uf[ni].v, (short)0, accu[mi][ni], false, false);
      }
    __syncthreads();
  }

#pragma unroll
  for (int mi = 0; mi < 2; mi++)
#pragma unroll
    for (int ni = 0; ni < 4; ni++)
#pragma unroll
      for (int r = 0; r < 8; r++) {
        int row = bm + wm + mi * 16 + half * 8 + r;
        int col = bn + wn + ni * 16 + l16;
        float g = accg[mi][ni][r];
        float u = accu[mi][ni][r];
        float s = g / (1.f + __expf(-g));
        out[(size_t)row * N + col] = f2bf(s * u);
      }
}

// ---------------------------------------------------------------------------
// RoPE on q/k (f32 in, bf16 out, head-major layouts) + V transpose to [d][S]
// one block per (b,s) token, 256 threads
// ---------------------------------------------------------------------------
__global__ __launch_bounds__(256)
void rope_pack(const float* __restrict__ qf, const float* __restrict__ kf,
               const float* __restrict__ vf, const int* __restrict__ positions,
               bf16_t* __restrict__ qb, bf16_t* __restrict__ kb,
               bf16_t* __restrict__ vtb) {
  int bs = blockIdx.x;
  int b = bs / Ss, s = bs % Ss;
  float pos = (float)positions[s];
  int t = threadIdx.x;
  const float LN_THETA = 9.210340371976184f;   // ln(10000)

  // q heads
  for (int j = t; j < Hh * 64; j += 256) {
    int h = j >> 6, i = j & 63;
    float invf = __expf(-((2.f * i) / 128.f) * LN_THETA);
    float sn, cs;
    __sincosf(pos * invf, &sn, &cs);
    const float* qr = qf + (size_t)bs * (Hh * Dh) + h * Dh;
    bf16_t* qo = qb + ((size_t)(b * Hh + h) * Ss + s) * Dh;
    float q1 = qr[i], q2 = qr[i + 64];
    qo[i]      = f2bf(q1 * cs - q2 * sn);
    qo[i + 64] = f2bf(q2 * cs + q1 * sn);
  }
  // k heads
  for (int j = t; j < KVh * 64; j += 256) {
    int h = j >> 6, i = j & 63;
    float invf = __expf(-((2.f * i) / 128.f) * LN_THETA);
    float sn, cs;
    __sincosf(pos * invf, &sn, &cs);
    const float* kr = kf + (size_t)bs * (KVh * Dh) + h * Dh;
    bf16_t* ko = kb + ((size_t)(b * KVh + h) * Ss + s) * Dh;
    float k1 = kr[i], k2 = kr[i + 64];
    ko[i]      = f2bf(k1 * cs - k2 * sn);
    ko[i + 64] = f2bf(k2 * cs + k1 * sn);
  }
  // v transpose: [b][kv][d][S]
  for (int j = t; j < KVh * Dh; j += 256) {
    int h = j >> 7, d = j & 127;
    float v = vf[(size_t)bs * (KVh * Dh) + h * Dh + d];
    vtb[((size_t)(b * KVh + h) * Dh + d) * Ss + s] = f2bf(v);
  }
}

// ---------------------------------------------------------------------------
// Flash attention (transposed formulation), causal, GQA 4:1
// grid (B*H, S/64), 128 threads = 4 waves; each wave owns 16 q rows
// ---------------------------------------------------------------------------
__global__ __launch_bounds__(128)
void attn_kernel(const bf16_t* __restrict__ qb, const bf16_t* __restrict__ kb,
                 const bf16_t* __restrict__ vtb, bf16_t* __restrict__ attnb) {
  int bh = blockIdx.x;
  int b = bh / Hh, h = bh % Hh, kvh = h / (Hh / KVh);
  int wave = threadIdx.x >> 5, lane = threadIdx.x & 31;
  int half = lane >> 4, l16 = lane & 15;
  int qBase = blockIdx.y * 64 + wave * 16;
  int qrow = qBase + l16;

  const bf16_t* qp = qb  + (size_t)(b * Hh  + h)   * Ss * Dh;
  const bf16_t* kp = kb  + (size_t)(b * KVh + kvh) * Ss * Dh;
  const bf16_t* vp = vtb + (size_t)(b * KVh + kvh) * Dh * Ss;

  // Q^T as WMMA B operand: 4 chunks of K(d)=32
  Frag qfr[4];
#pragma unroll
  for (int c = 0; c < 4; c++) {
    size_t off = (size_t)qrow * Dh + c * 32 + half * 8;
    qfr[c].u[0] = *(const uint4*)(qp + off);
    qfr[c].u[1] = *(const uint4*)(qp + off + 16);
  }

  v8f Oacc[8] = {};
  float m = -1e30f, l = 0.f;
  const float scale = 0.08838834764831845f;   // 1/sqrt(128)

  int nBlocks = (qBase + 16 + 31) >> 5;
  for (int blk = 0; blk < nBlocks; blk++) {
    int kvBase = blk * 32;
    v8f st0 = {}, st1 = {};
#pragma unroll
    for (int c = 0; c < 4; c++) {
      Frag kf0, kf1;
      size_t off0 = (size_t)(kvBase + l16) * Dh + c * 32 + half * 8;
      size_t off1 = (size_t)(kvBase + 16 + l16) * Dh + c * 32 + half * 8;
      kf0.u[0] = *(const uint4*)(kp + off0);
      kf0.u[1] = *(const uint4*)(kp + off0 + 16);
      kf1.u[0] = *(const uint4*)(kp + off1);
      kf1.u[1] = *(const uint4*)(kp + off1 + 16);
      st0 = __builtin_amdgcn_wmma_f32_16x16x32_bf16(false, kf0.v, false, qfr[c].v,
                                                    (short)0, st0, false, false);
      st1 = __builtin_amdgcn_wmma_f32_16x16x32_bf16(false, kf1.v, false, qfr[c].v,
                                                    (short)0, st1, false, false);
    }

    float p0[8], p1[8];
    float mloc = -1e30f;
#pragma unroll
    for (int r = 0; r < 8; r++) {
      int kv0 = kvBase + half * 8 + r;
      int kv1 = kv0 + 16;
      float x0 = (kv0 <= qrow) ? st0[r] * scale : -1e30f;
      float x1 = (kv1 <= qrow) ? st1[r] * scale : -1e30f;
      p0[r] = x0; p1[r] = x1;
      mloc = fmaxf(mloc, fmaxf(x0, x1));
    }
    mloc = fmaxf(mloc, __shfl_xor(mloc, 16));
    float mnew = fmaxf(m, mloc);
    float alpha = __expf(m - mnew);
    float lsum = 0.f;
#pragma unroll
    for (int r = 0; r < 8; r++) {
      p0[r] = __expf(p0[r] - mnew);
      p1[r] = __expf(p1[r] - mnew);
      lsum += p0[r] + p1[r];
    }
    lsum += __shfl_xor(lsum, 16);
    l = l * alpha + lsum;
    m = mnew;
#pragma unroll
    for (int mt = 0; mt < 8; mt++)
#pragma unroll
      for (int r = 0; r < 8; r++) Oacc[mt][r] *= alpha;

    // build P^T fragment (B operand for V^T @ P^T)
    Frag pf;
#pragma unroll
    for (int r = 0; r < 8; r++) {
      float b0 = __shfl_xor(p0[r], 16);
      float b1 = __shfl_xor(p1[r], 16);
      float elo = half ? b1 : p0[r];
      float ehi = half ? p1[r] : b0;
      pf.h[r]     = f2bf(elo);
      pf.h[r + 8] = f2bf(ehi);
    }

#pragma unroll
    for (int mt = 0; mt < 8; mt++) {
      Frag vfr;
      size_t off = (size_t)(mt * 16 + l16) * Ss + kvBase + half * 16;
      vfr.u[0] = *(const uint4*)(vp + off);
      vfr.u[1] = *(const uint4*)(vp + off + 8);
      Oacc[mt] = __builtin_amdgcn_wmma_f32_16x16x32_bf16(
          false, vfr.v, false, pf.v, (short)0, Oacc[mt], false, false);
    }
  }

  float rinv = 1.f / l;
  bf16_t* outp = attnb + ((size_t)(b * Ss + qrow)) * (Hh * Dh) + h * Dh;
#pragma unroll
  for (int mt = 0; mt < 8; mt++)
#pragma unroll
    for (int r = 0; r < 8; r++) {
      int d = mt * 16 + half * 8 + r;
      outp[d] = f2bf(Oacc[mt][r] * rinv);
    }
}

// ---------------------------------------------------------------------------
// Host-side orchestration
// ---------------------------------------------------------------------------
extern "C" void kernel_launch(void* const* d_in, const int* in_sizes, int n_in,
                              void* d_out, int out_size, void* d_ws, size_t ws_size,
                              hipStream_t stream) {
  const float* hidden = (const float*)d_in[0];
  const int*   pos    = (const int*)d_in[1];
  const float* wq     = (const float*)d_in[2];
  const float* wk     = (const float*)d_in[3];
  const float* wv     = (const float*)d_in[4];
  const float* wo     = (const float*)d_in[5];
  const float* wg     = (const float*)d_in[6];
  const float* wu     = (const float*)d_in[7];
  const float* wd     = (const float*)d_in[8];
  const float* ln1    = (const float*)d_in[9];
  const float* ln2    = (const float*)d_in[10];
  float* out = (float*)d_out;

  char* base = (char*)d_ws;
  size_t off = 0;
  auto alloc = [&](size_t bytes) -> void* {
    void* p = base + off;
    off += (bytes + 255) & ~(size_t)255;
    return p;
  };
  bf16_t* wqT  = (bf16_t*)alloc((size_t)Dm * Dm * 2);            // [2048][2048]
  bf16_t* wkT  = (bf16_t*)alloc((size_t)(KVh * Dh) * Dm * 2);    // [512][2048]
  bf16_t* wvT  = (bf16_t*)alloc((size_t)(KVh * Dh) * Dm * 2);
  bf16_t* woT  = (bf16_t*)alloc((size_t)Dm * Dm * 2);
  bf16_t* wgT  = (bf16_t*)alloc((size_t)Dff * Dm * 2);           // [8192][2048]
  bf16_t* wuT  = (bf16_t*)alloc((size_t)Dff * Dm * 2);
  bf16_t* wdT  = (bf16_t*)alloc((size_t)Dm * Dff * 2);           // [2048][8192]
  bf16_t* hn   = (bf16_t*)alloc((size_t)Mtot * Dm * 2);
  float*  qf   = (float*)alloc((size_t)Mtot * Dm * 4);
  float*  kf   = (float*)alloc((size_t)Mtot * (KVh * Dh) * 4);
  float*  vf   = (float*)alloc((size_t)Mtot * (KVh * Dh) * 4);
  bf16_t* qbuf = (bf16_t*)alloc((size_t)Mtot * Dm * 2);          // [b][h][s][d]
  bf16_t* kbuf = (bf16_t*)alloc((size_t)Mtot * (KVh * Dh) * 2);  // [b][kv][s][d]
  bf16_t* vtb  = (bf16_t*)alloc((size_t)Mtot * (KVh * Dh) * 2);  // [b][kv][d][s]
  bf16_t* attnb= (bf16_t*)alloc((size_t)Mtot * Dm * 2);
  float*  x1   = (float*)alloc((size_t)Mtot * Dm * 4);
  bf16_t* h2   = (bf16_t*)alloc((size_t)Mtot * Dm * 2);
  bf16_t* hff  = (bf16_t*)alloc((size_t)Mtot * Dff * 2);
  (void)ws_size; (void)in_sizes; (void)n_in; (void)out_size;

  dim3 tb(32, 8);
  transpose_to_bf16<<<dim3(Dm / 32,  Dm / 32),  tb, 0, stream>>>(wq, wqT, Dm, Dm);
  transpose_to_bf16<<<dim3(16,       Dm / 32),  tb, 0, stream>>>(wk, wkT, Dm, KVh * Dh);
  transpose_to_bf16<<<dim3(16,       Dm / 32),  tb, 0, stream>>>(wv, wvT, Dm, KVh * Dh);
  transpose_to_bf16<<<dim3(Dm / 32,  Dm / 32),  tb, 0, stream>>>(wo, woT, Dm, Dm);
  transpose_to_bf16<<<dim3(Dff / 32, Dm / 32),  tb, 0, stream>>>(wg, wgT, Dm, Dff);
  transpose_to_bf16<<<dim3(Dff / 32, Dm / 32),  tb, 0, stream>>>(wu, wuT, Dm, Dff);
  transpose_to_bf16<<<dim3(Dm / 32,  Dff / 32), tb, 0, stream>>>(wd, wdT, Dff, Dm);

  rmsnorm_bf16<<<Mtot, 256, 0, stream>>>(hidden, ln1, hn, Dm);

  gemm_bf16<<<dim3(Dm / 128, Mtot / 128), 256, 0, stream>>>(
      hn, wqT, qf, nullptr, Mtot, Dm, Dm);
  gemm_bf16<<<dim3((KVh * Dh) / 128, Mtot / 128), 256, 0, stream>>>(
      hn, wkT, kf, nullptr, Mtot, KVh * Dh, Dm);
  gemm_bf16<<<dim3((KVh * Dh) / 128, Mtot / 128), 256, 0, stream>>>(
      hn, wvT, vf, nullptr, Mtot, KVh * Dh, Dm);

  rope_pack<<<Mtot, 256, 0, stream>>>(qf, kf, vf, pos, qbuf, kbuf, vtb);

  attn_kernel<<<dim3(Bb * Hh, Ss / 64), 128, 0, stream>>>(qbuf, kbuf, vtb, attnb);

  gemm_bf16<<<dim3(Dm / 128, Mtot / 128), 256, 0, stream>>>(
      attnb, woT, x1, hidden, Mtot, Dm, Dm);

  rmsnorm_bf16<<<Mtot, 256, 0, stream>>>(x1, ln2, h2, Dm);

  gemm_gateup<<<dim3(Dff / 128, Mtot / 128), 256, 0, stream>>>(
      h2, wgT, wuT, hff, Mtot, Dff, Dm);

  gemm_bf16<<<dim3(Dm / 128, Mtot / 128), 256, 0, stream>>>(
      hff, wdT, out, x1, Mtot, Dm, Dff);
}
